// GCLSTM2_5076651344196
// MI455X (gfx1250) — matrix operand order repaired and checked by reading, not compiled, not run on hardware
//
#include <hip/hip_runtime.h>
#include <math.h>

// ---------------------------------------------------------------------------
// GC-LSTM (ChebConv K=3, sym norm, lambda_max=2) for MI455X / gfx1250.
//   T=8, N=50000, E=800000, D_IN=D_OUT=64, K=3
// Design: per step ->
//   [deg] -> [norm] -> [SpMM Tx1=S*H] -> [SpMM Tx2=2*S*Tx1 - H]
//   -> fused WMMA gate kernel: z = [x|H|Tx1|Tx2](N x 256) @ W_all(256 x 256)
//      + LSTM pointwise (i,f,c~,o with peepholes), in-place H/C update.
// Dense math uses v_wmma_f32_16x16x32_bf16 (bf16 A/B, f32 accumulate).
// ---------------------------------------------------------------------------

typedef __attribute__((ext_vector_type(16))) __bf16 v16bf;
typedef __attribute__((ext_vector_type(8)))  float  v8f;

#define TT 8
#define NN 50000
#define EE 800000

__device__ __forceinline__ unsigned short f2bf(float f) {
  return __builtin_bit_cast(unsigned short, (__bf16)f);
}

// ---------------------------------------------------------------------------
// Pack W_all (256x256 f32, rows = K over [x|H|Tx1|Tx2], cols = 4 gates x 64)
// into the exact per-lane WMMA B-matrix (32x16 bf16) layout:
//   index = (((chunk*16)+tile)*32 + lane)*16 + i,  value = W_all[K][col]
//   lane<16 : col = tile*16+lane,      K = chunk*32 + (i<8 ? i : 8 + i)
//   lane>=16: col = tile*16+(lane-16), K = chunk*32 + 8 + (i<8 ? i : 8 + i)
// W_all rows: [0,64): Wx[g];  [64,128): Th[g][0];  [128,192): Th[g][1];
//             [192,256): Th[g][2]   with col = g*64 + j.
// ---------------------------------------------------------------------------
__global__ __launch_bounds__(256) void k_pack_weights(
    const float* __restrict__ Wx, const float* __restrict__ Th,
    unsigned short* __restrict__ Wp)
{
  const int gtid = blockIdx.x * 256 + threadIdx.x;  // 0..4095
  const int c    = gtid >> 9;          // chunk 0..7
  const int t    = (gtid >> 5) & 15;   // tile  0..15
  const int lane = gtid & 31;
  const int hi16 = (lane & 16) ? 16 : 0;
  const int col  = t * 16 + (lane & 15);
  const int g = col >> 6, j = col & 63;
#pragma unroll
  for (int i = 0; i < 16; ++i) {
    const int K = c * 32 + (hi16 >> 1) + (i < 8 ? i : 8 + i);
    float v;
    if (K < 64)       v = Wx[(size_t)g * 4096 + (size_t)K * 64 + j];
    else if (K < 128) v = Th[((size_t)g * 3 + 0) * 4096 + (size_t)(K - 64)  * 64 + j];
    else if (K < 192) v = Th[((size_t)g * 3 + 1) * 4096 + (size_t)(K - 128) * 64 + j];
    else              v = Th[((size_t)g * 3 + 2) * 4096 + (size_t)(K - 192) * 64 + j];
    Wp[(size_t)gtid * 16 + i] = f2bf(v);
  }
}

__global__ __launch_bounds__(256) void k_init_state(float* __restrict__ H,
                                                    float* __restrict__ C) {
  const int idx = blockIdx.x * 256 + threadIdx.x;
  if (idx < NN * 64) { H[idx] = 0.0f; C[idx] = 0.0f; }
}

// Tx1 = 0, Tx2 = -H  (so SpMM2 with scale 2 yields Tx2 = 2*S*Tx1 - H), deg = 0
__global__ __launch_bounds__(256) void k_step_init(
    float* __restrict__ Tx1, float* __restrict__ Tx2,
    const float* __restrict__ H, float* __restrict__ deg)
{
  const int idx = blockIdx.x * 256 + threadIdx.x;
  if (idx < NN * 64) { Tx1[idx] = 0.0f; Tx2[idx] = -H[idx]; }
  if (idx < NN) deg[idx] = 0.0f;
}

__global__ __launch_bounds__(256) void k_deg(const int* __restrict__ src,
                                             const float* __restrict__ w,
                                             float* __restrict__ deg) {
  const int e = blockIdx.x * 256 + threadIdx.x;
  if (e < EE) atomicAdd(&deg[src[e]], w[e]);
}

__global__ __launch_bounds__(256) void k_norm(
    const int* __restrict__ src, const int* __restrict__ dst,
    const float* __restrict__ w, const float* __restrict__ deg,
    float* __restrict__ nrm)
{
  const int e = blockIdx.x * 256 + threadIdx.x;
  if (e >= EE) return;
  const float ds = deg[src[e]], dd = deg[dst[e]];
  const float a = (ds > 0.0f) ? rsqrtf(fmaxf(ds, 1e-12f)) : 0.0f;
  const float c = (dd > 0.0f) ? rsqrtf(fmaxf(dd, 1e-12f)) : 0.0f;
  nrm[e] = -a * w[e] * c;
}

// out[dst] += scale*norm[e] * in[src]  over 64 features; 16 lanes/edge, float4
__global__ __launch_bounds__(256) void k_spmm(
    const int* __restrict__ src, const int* __restrict__ dst,
    const float* __restrict__ nrm, const float* __restrict__ in,
    float* __restrict__ out, float scale)
{
  const long long gtid = (long long)blockIdx.x * 256 + threadIdx.x;
  const int e = (int)(gtid >> 4);
  if (e >= EE) return;
  const float s = nrm[e] * scale;
  if (s == 0.0f) return;
  const int part = (int)(gtid & 15);
  const float4 v = *(const float4*)(in + (size_t)src[e] * 64 + part * 4);
  float* o = out + (size_t)dst[e] * 64 + part * 4;
  atomicAdd(o + 0, s * v.x);
  atomicAdd(o + 1, s * v.y);
  atomicAdd(o + 2, s * v.z);
  atomicAdd(o + 3, s * v.w);
}

// ---------------------------------------------------------------------------
// Fused gate kernel: 16 nodes/block, 8 waves, each wave -> 2 column tiles.
// z(16x256) = A(16x256) @ W_all(256x256), A = [x|H|Tx1|Tx2] rows of this block.
// Then the LSTM cell update (in place on H,C; safe: block-private rows,
// all A reads complete before the barrier).
// ---------------------------------------------------------------------------
__global__ __launch_bounds__(256) void k_gate(
    const float* __restrict__ x_t, float* __restrict__ H, float* __restrict__ C,
    const float* __restrict__ Tx1, const float* __restrict__ Tx2,
    const unsigned short* __restrict__ Wp,
    const float* __restrict__ b, const float* __restrict__ wc)
{
  __shared__ float zs[16 * 256];
  const int tid  = threadIdx.x;
  const int wave = tid >> 5;
  const int lane = tid & 31;
  const int lc   = lane & 15;
  const int hi   = lane >> 4;
  const int rowBase = blockIdx.x * 16;
  const size_t rowOff = (size_t)(rowBase + lc) * 64;

  const float* ap0 = x_t + rowOff;
  const float* ap1 = H   + rowOff;
  const float* ap2 = Tx1 + rowOff;
  const float* ap3 = Tx2 + rowOff;

  const v16bf* Bp = (const v16bf*)Wp;
  const int t0 = wave, t1 = wave + 8;

  v8f acc0 = {};
  v8f acc1 = {};

#pragma unroll
  for (int c = 0; c < 8; ++c) {
    const float* p = (c < 2) ? ap0 : (c < 4) ? ap1 : (c < 6) ? ap2 : ap3;
    const int off = (c & 1) * 32;
    const int s0 = off + hi * 8;        // lane<16: K 0-7   ; lane>=16: K 8-15
    const int s1 = off + 16 + hi * 8;   // lane<16: K 16-23 ; lane>=16: K 24-31
    const float4 q0 = *(const float4*)(p + s0);
    const float4 q1 = *(const float4*)(p + s0 + 4);
    const float4 q2 = *(const float4*)(p + s1);
    const float4 q3 = *(const float4*)(p + s1 + 4);
    v16bf a;
    a[0]=(__bf16)q0.x;  a[1]=(__bf16)q0.y;  a[2]=(__bf16)q0.z;  a[3]=(__bf16)q0.w;
    a[4]=(__bf16)q1.x;  a[5]=(__bf16)q1.y;  a[6]=(__bf16)q1.z;  a[7]=(__bf16)q1.w;
    a[8]=(__bf16)q2.x;  a[9]=(__bf16)q2.y;  a[10]=(__bf16)q2.z; a[11]=(__bf16)q2.w;
    a[12]=(__bf16)q3.x; a[13]=(__bf16)q3.y; a[14]=(__bf16)q3.z; a[15]=(__bf16)q3.w;

    const v16bf b0 = Bp[(c * 16 + t0) * 32 + lane];
    const v16bf b1 = Bp[(c * 16 + t1) * 32 + lane];
    acc0 = __builtin_amdgcn_wmma_f32_16x16x32_bf16(false, a, false, b0,
                                                   (short)0, acc0, false, false);
    acc1 = __builtin_amdgcn_wmma_f32_16x16x32_bf16(false, a, false, b1,
                                                   (short)0, acc1, false, false);
  }

  // C/D layout: VGPR j -> row (j + 8*hi), col = lane&15 within tile
#pragma unroll
  for (int j = 0; j < 8; ++j) {
    const int m = j + 8 * hi;
    zs[m * 256 + t0 * 16 + lc] = acc0[j];
    zs[m * 256 + t1 * 16 + lc] = acc1[j];
  }
  __syncthreads();

  // LSTM pointwise: 16 nodes x 64 features = 1024 elements, 4 per thread
#pragma unroll
  for (int p = 0; p < 4; ++p) {
    const int e = p * 256 + tid;
    const int n = e >> 6;
    const int j = e & 63;
    const size_t g = (size_t)(rowBase + n) * 64 + j;
    const float z0 = zs[n * 256 + j]       + b[j];
    const float z1 = zs[n * 256 + 64 + j]  + b[64 + j];
    const float z2 = zs[n * 256 + 128 + j] + b[128 + j];
    const float z3 = zs[n * 256 + 192 + j] + b[192 + j];
    const float c_old = C[g];
    const float ig = 1.0f / (1.0f + __expf(-(z0 + wc[j] * c_old)));
    const float fg = 1.0f / (1.0f + __expf(-(z1 + wc[64 + j] * c_old)));
    const float cn = fg * c_old + ig * tanhf(z2);
    const float og = 1.0f / (1.0f + __expf(-(z3 + wc[128 + j] * cn)));
    H[g] = og * tanhf(cn);
    C[g] = cn;
  }
}

__global__ __launch_bounds__(256) void k_relu(const float* __restrict__ H,
                                              float* __restrict__ out) {
  const int idx = blockIdx.x * 256 + threadIdx.x;
  if (idx < NN * 64) out[idx] = fmaxf(H[idx], 0.0f);
}

// ---------------------------------------------------------------------------
extern "C" void kernel_launch(void* const* d_in, const int* in_sizes, int n_in,
                              void* d_out, int out_size, void* d_ws, size_t ws_size,
                              hipStream_t stream) {
  (void)in_sizes; (void)n_in; (void)out_size; (void)ws_size;
  const float* x_seq  = (const float*)d_in[0];  // [T,N,64]
  const float* ew_seq = (const float*)d_in[1];  // [T,E]
  const float* Wx     = (const float*)d_in[2];  // [4,64,64]
  const float* Th     = (const float*)d_in[3];  // [4,3,64,64]
  const float* b      = (const float*)d_in[4];  // [4,64]
  const float* wc     = (const float*)d_in[5];  // [3,64]
  const int*   ei_seq = (const int*)d_in[6];    // [T,2,E]
  float* out = (float*)d_out;

  const size_t N64 = (size_t)NN * 64;
  float* H   = (float*)d_ws;          // 12.8 MB each
  float* C   = H   + N64;
  float* Tx1 = C   + N64;
  float* Tx2 = Tx1 + N64;
  float* deg = Tx2 + N64;             // N floats
  float* nrm = deg + NN;              // E floats
  unsigned short* Wp = (unsigned short*)(nrm + EE);  // 64K bf16 (128 KB)

  const int gN64 = (int)((N64 + 255) / 256);

  k_pack_weights<<<16, 256, 0, stream>>>(Wx, Th, Wp);
  k_init_state<<<gN64, 256, 0, stream>>>(H, C);

  for (int t = 0; t < TT; ++t) {
    const float* x_t = x_seq + (size_t)t * N64;
    const int*   src = ei_seq + (size_t)t * 2 * EE;
    const int*   dst = src + EE;
    const float* w   = ew_seq + (size_t)t * EE;

    k_step_init<<<gN64, 256, 0, stream>>>(Tx1, Tx2, H, deg);
    k_deg <<<EE / 256, 256, 0, stream>>>(src, w, deg);
    k_norm<<<EE / 256, 256, 0, stream>>>(src, dst, w, deg, nrm);
    k_spmm<<<(EE * 16) / 256, 256, 0, stream>>>(src, dst, nrm, H,   Tx1, 1.0f);
    k_spmm<<<(EE * 16) / 256, 256, 0, stream>>>(src, dst, nrm, Tx1, Tx2, 2.0f);
    k_gate<<<NN / 16, 256, 0, stream>>>(x_t, H, C, Tx1, Tx2, Wp, b, wc);
  }

  k_relu<<<gN64, 256, 0, stream>>>(H, out);
}
